// LSTMcell_27736898798344
// MI455X (gfx1250) — matrix operand order repaired and checked by reading, not compiled, not run on hardware
//
#include <hip/hip_runtime.h>
#include <stdint.h>

typedef _Float16 v16h __attribute__((ext_vector_type(16)));
typedef _Float16 v8h  __attribute__((ext_vector_type(8)));
typedef _Float16 v4h  __attribute__((ext_vector_type(4)));
typedef float    v8f  __attribute__((ext_vector_type(8)));

#define BATCH   8192
#define HID     1024
#define KDIM    2048
#define MT      128              // rows per workgroup
#define NT      64               // cols per gate per workgroup (effective N = 256)
#define KT      64               // K per stage
#define NSTAGE  (KDIM / KT)      // 32
#define LSTRIDE 72               // f16 elements per LDS row (64 + 8 pad)
#define ROWB    (LSTRIDE * 2)    // 144 bytes, 16B aligned, conflict-free
#define ABUFB   (MT * ROWB)      // 18432 B per A buffer
#define BBUFB   (4 * NT * ROWB)  // 36864 B per B buffer
#define B_OFF   (2 * ABUFB)      // B region after double-buffered A
#define LDS_TOTAL (2 * ABUFB + 2 * BBUFB)   // 110592 B

__device__ __forceinline__ float fast_sigmoid(float x) {
    return 1.0f / (1.0f + __expf(-x));
}
__device__ __forceinline__ float fast_tanh(float x) {
    return 2.0f / (1.0f + __expf(-2.0f * x)) - 1.0f;
}

// ---------- pre-pass 1: A' = f16(concat(x, h))  [8192][2048] ----------
__global__ __launch_bounds__(256) void prep_a(
    const float* __restrict__ x, const float* __restrict__ h,
    _Float16* __restrict__ A)
{
    const int idx = (blockIdx.x * 256 + threadIdx.x) * 8;   // 8 elems/thread
    const int row = idx >> 11;
    const int k   = idx & 2047;
    const float* src = (k < 1024) ? (x + row * 1024 + k) : (h + row * 1024 + (k - 1024));
    const float4 f0 = *(const float4*)src;
    const float4 f1 = *(const float4*)(src + 4);
    v8h o;
    o[0] = (_Float16)f0.x; o[1] = (_Float16)f0.y; o[2] = (_Float16)f0.z; o[3] = (_Float16)f0.w;
    o[4] = (_Float16)f1.x; o[5] = (_Float16)f1.y; o[6] = (_Float16)f1.z; o[7] = (_Float16)f1.w;
    *(v8h*)(A + idx) = o;
}

// ---------- pre-pass 2: W'[g][n][k] = f16(Wg[k][n])  (transpose+convert) ----------
__global__ __launch_bounds__(256) void prep_w(
    const float* __restrict__ Wf, const float* __restrict__ Wi,
    const float* __restrict__ Wc, const float* __restrict__ Wo,
    _Float16* __restrict__ Wt)
{
    __shared__ float tile[32][33];
    const int g  = blockIdx.z;
    const float* W = (g == 0) ? Wf : (g == 1) ? Wi : (g == 2) ? Wc : Wo;
    const int k0 = blockIdx.x * 32;
    const int n0 = blockIdx.y * 32;
    const int tr = threadIdx.x >> 3;          // 0..31
    const int tc = (threadIdx.x & 7) * 4;     // 0,4,..,28
    const float4 v = *(const float4*)(W + (k0 + tr) * HID + n0 + tc);
    tile[tr][tc + 0] = v.x; tile[tr][tc + 1] = v.y;
    tile[tr][tc + 2] = v.z; tile[tr][tc + 3] = v.w;
    __syncthreads();
    v4h o;
    o[0] = (_Float16)tile[tc + 0][tr];
    o[1] = (_Float16)tile[tc + 1][tr];
    o[2] = (_Float16)tile[tc + 2][tr];
    o[3] = (_Float16)tile[tc + 3][tr];
    *(v4h*)(Wt + (size_t)(g * HID + n0 + tr) * KDIM + k0 + tc) = o;
}

// ---------- main: fused 4-gate GEMM + LSTM epilogue ----------
__global__ __launch_bounds__(256) void lstm_gemm(
    const _Float16* __restrict__ A, const _Float16* __restrict__ Wt,
    const float* __restrict__ cprev,
    const float* __restrict__ bf, const float* __restrict__ bi,
    const float* __restrict__ bc, const float* __restrict__ bo,
    float* __restrict__ out)
{
    extern __shared__ char smem[];
    const uint32_t ldsBase = (uint32_t)(uintptr_t)smem;   // low 32 bits = LDS byte offset

    const int tid   = threadIdx.x;
    const int wave  = tid >> 5;
    const int lane  = tid & 31;
    const int nl    = lane & 15;
    const int khalf = lane >> 4;
    const int tileN = blockIdx.x * NT;
    const int tileM = blockIdx.y * MT;

    // async staging mapping: 16B chunks, all contiguous copies (data already f16)
    const int ar = tid >> 3;          // A: row group 0..31 (4 passes of 32 rows)
    const int ac = tid & 7;           // A: chunk within 64-f16 row
    const int bg = tid >> 6;          // B: gate (wave-uniform)
    const int bq = tid & 63;
    const int bn0 = bq >> 3;          // B: n group 0..7 (8 passes of 8 rows)
    const int bcc = bq & 7;           // B: chunk within row

    auto stage_async = [&](int s, int buf) {
        const int kbase = s * KT;
#pragma unroll
        for (int p = 0; p < 4; ++p) {
            const int row = ar + p * 32;
            const uint64_t ga = (uint64_t)(uintptr_t)(A + (size_t)(tileM + row) * KDIM + kbase) + ac * 16;
            const uint32_t la = ldsBase + buf * ABUFB + row * ROWB + ac * 16;
            asm volatile("global_load_async_to_lds_b128 %0, %1, off"
                         :: "v"(la), "v"(ga) : "memory");
        }
#pragma unroll
        for (int p = 0; p < 8; ++p) {
            const int n = bn0 + p * 8;
            const uint64_t gb = (uint64_t)(uintptr_t)
                (Wt + (size_t)(bg * HID + tileN + n) * KDIM + kbase) + bcc * 16;
            const uint32_t lb = ldsBase + B_OFF + buf * BBUFB + (bg * NT + n) * ROWB + bcc * 16;
            asm volatile("global_load_async_to_lds_b128 %0, %1, off"
                         :: "v"(lb), "v"(gb) : "memory");
        }
    };

    v8f acc[4][4];
#pragma unroll
    for (int g = 0; g < 4; ++g)
#pragma unroll
        for (int n = 0; n < 4; ++n)
            acc[g][n] = (v8f){};

    const int mrow = wave * 16 + nl;

    // B frag 32x16: lane = col, lane-half selects K 0-15 / 16-31
    auto loadB = [&](int buf, int i) -> v16h {
        const int kb = i >> 4;
        const int g  = (i >> 2) & 3;
        const int nt = i & 3;
        const _Float16* bp = (const _Float16*)
            (smem + B_OFF + buf * BBUFB + (g * NT + nt * 16 + nl) * ROWB)
            + kb * 32 + khalf * 16;
        const v8h blo = *(const v8h*)bp;
        const v8h bhi = *(const v8h*)(bp + 8);
        return __builtin_shufflevector(blo, bhi,
            0,1,2,3,4,5,6,7,8,9,10,11,12,13,14,15);
    };

    auto compute = [&](int buf) {
        // hoist both A frags (16x32: lanes 0-15 -> K 0..15, 16-31 -> K 16..31)
        v16h af[2];
#pragma unroll
        for (int kb = 0; kb < 2; ++kb) {
            const _Float16* ap = (const _Float16*)(smem + buf * ABUFB + mrow * ROWB)
                                 + kb * 32 + khalf * 16;
            const v8h alo = *(const v8h*)ap;
            const v8h ahi = *(const v8h*)(ap + 8);
            af[kb] = __builtin_shufflevector(alo, ahi,
                0,1,2,3,4,5,6,7,8,9,10,11,12,13,14,15);
        }
        // software-pipelined B frags: load i+1 before WMMA(i) so LDS latency
        // hides behind the matrix pipe (2-deep dscnt instead of full waits)
        v16h bcur = loadB(buf, 0);
#pragma unroll
        for (int i = 0; i < 32; ++i) {
            v16h bnext;
            if (i + 1 < 32) bnext = loadB(buf, i + 1);
            const int g  = (i >> 2) & 3;
            const int nt = i & 3;
            acc[g][nt] = __builtin_amdgcn_wmma_f32_16x16x32_f16(
                false, af[i >> 4], false, bcur, (short)0, acc[g][nt], false, false);
            bcur = bnext;
        }
    };

    // ---- pipeline: async global->LDS double buffering, 1 barrier/stage ----
    stage_async(0, 0);
    asm volatile("s_wait_asynccnt 0x0" ::: "memory");
    __syncthreads();
#pragma unroll 1
    for (int s = 0; s < NSTAGE; ++s) {
        if (s + 1 < NSTAGE) stage_async(s + 1, (s + 1) & 1);
        compute(s & 1);
        if (s + 1 < NSTAGE) {
            asm volatile("s_wait_asynccnt 0x0" ::: "memory");
            __syncthreads();
        }
    }

    // ---- fused LSTM epilogue ----
    const int mhalf = lane >> 4;
#pragma unroll
    for (int nt = 0; nt < 4; ++nt) {
        const int col = tileN + nt * 16 + nl;
        const float bfv = bf[col], biv = bi[col], bcv = bc[col], bov = bo[col];
#pragma unroll
        for (int v = 0; v < 8; ++v) {
            const int row = tileM + wave * 16 + mhalf * 8 + v;  // C/D layout: VGPR v -> M=v / v+8
            const float fg = fast_sigmoid(acc[0][nt][v] + bfv);
            const float ig = fast_sigmoid(acc[1][nt][v] + biv);
            const float cd = fast_tanh   (acc[2][nt][v] + bcv);
            const float og = fast_sigmoid(acc[3][nt][v] + bov);
            const float cold = cprev[row * HID + col];
            const float nc = fg * cold + ig * cd;
            const float nh = og * fast_tanh(nc);
            out[row * HID + col] = nh;                    // new_hidden
            out[BATCH * HID + row * HID + col] = nc;      // new_cell
        }
    }
}

extern "C" void kernel_launch(void* const* d_in, const int* in_sizes, int n_in,
                              void* d_out, int out_size, void* d_ws, size_t ws_size,
                              hipStream_t stream) {
    const float* xin   = (const float*)d_in[0];
    const float* hin   = (const float*)d_in[1];
    const float* cprev = (const float*)d_in[2];
    const float* Wf = (const float*)d_in[3];  const float* bf = (const float*)d_in[4];
    const float* Wi = (const float*)d_in[5];  const float* bi = (const float*)d_in[6];
    const float* Wc = (const float*)d_in[7];  const float* bc = (const float*)d_in[8];
    const float* Wo = (const float*)d_in[9];  const float* bo = (const float*)d_in[10];
    float* out = (float*)d_out;

    // workspace: A' (32 MB f16) then W' (16 MB f16)
    _Float16* Ah = (_Float16*)d_ws;
    _Float16* Wt = Ah + (size_t)BATCH * KDIM;

    prep_a<<<dim3(BATCH * KDIM / (256 * 8)), 256, 0, stream>>>(xin, hin, Ah);
    prep_w<<<dim3(KDIM / 32, HID / 32, 4), 256, 0, stream>>>(Wf, Wi, Wc, Wo, Wt);

    dim3 grid(HID / NT, BATCH / MT);   // (16, 64)
    lstm_gemm<<<grid, 256, LDS_TOTAL, stream>>>(
        Ah, Wt, cprev, bf, bi, bc, bo, out);
}
